// MultiInputGCN_88785563943603
// MI455X (gfx1250) — matrix operand order repaired and checked by reading, not compiled, not run on hardware
//
#include <hip/hip_runtime.h>
#include <hip/hip_bf16.h>

typedef __attribute__((ext_vector_type(16))) __bf16 v16bf;
typedef __attribute__((ext_vector_type(8)))  float  v8f;

// ---------------------------------------------------------------------------
// Small elementwise / setup kernels
// ---------------------------------------------------------------------------
__global__ void k_fill_f32(float* p, float v, int n) {
    int i = blockIdx.x * blockDim.x + threadIdx.x;
    if (i < n) p[i] = v;
}

__global__ void k_deg_atomic(const long long* __restrict__ tgt, float* deg, int E) {
    int e = blockIdx.x * blockDim.x + threadIdx.x;
    if (e < E) atomicAdd(&deg[(int)tgt[e]], 1.0f);
}

__global__ void k_rsqrt(const float* __restrict__ deg, float* dis, int n) {
    int i = blockIdx.x * blockDim.x + threadIdx.x;
    if (i < n) dis[i] = rsqrtf(deg[i]);
}

__global__ void k_f32_to_bf16(const float* __restrict__ in, __bf16* out, int n) {
    int i = blockIdx.x * blockDim.x + threadIdx.x;
    if (i < n) out[i] = (__bf16)in[i];
}

// W: [K][N] row-major f32  ->  Wt: [N][K] bf16 (so B fragments load contiguously)
__global__ void k_transpose_bf16(const float* __restrict__ W, __bf16* Wt, int K, int N) {
    int idx = blockIdx.x * blockDim.x + threadIdx.x;
    if (idx < K * N) {
        int kk = idx % K;
        int n  = idx / K;
        Wt[(long)n * K + kk] = (__bf16)W[(long)kk * N + n];
    }
}

// ---------------------------------------------------------------------------
// Generic bf16 WMMA GEMM: C[M,N] = A[M,K] @ Bt[N,K]^T (+bias, +relu)
// A row-major bf16 (streamed from HBM), Bt = B transposed (N-major) bf16,
// staged through LDS in 256-wide K chunks shared by all 4 waves of the block.
// One wave computes a 16x64 strip: 1 A-fragment feeds 4 WMMAs per K-step;
// all 4 B-fragments occupy distinct VGPRs so one wait covers 4 matrix ops.
// M % 16 == 0 via wave-uniform 'active' flag; K % 32 == 0; N % 64 == 0.
// ---------------------------------------------------------------------------
#define GEMM_KC 256                         // K elements staged per chunk
#define GEMM_LDSROW (GEMM_KC + 8)           // +8 bf16 pad: 4-bank advance/row

__global__ void __launch_bounds__(128)
k_gemm_bf16(const __bf16* __restrict__ A, const __bf16* __restrict__ Bt,
            const float* __restrict__ bias, float* __restrict__ C,
            int M, int K, int N, int doBias, int doRelu)
{
    __shared__ __bf16 sB[64 * GEMM_LDSROW];             // 33 KB

    const int tid  = threadIdx.x;
    const int lane = tid & 31;
    const int wid  = tid >> 5;
    const int n0   = blockIdx.y * 64;
    const int m0   = (blockIdx.x * 4 + wid) * 16;
    const bool active = (m0 < M);                       // wave-uniform
    const int half = lane >> 4;                         // 0 | 1
    const int l16  = lane & 15;

    v8f acc[4];
    #pragma unroll
    for (int t = 0; t < 4; ++t)
        #pragma unroll
        for (int r = 0; r < 8; ++r) acc[t][r] = 0.0f;

    // A-fragment layout (16-bit 16x32): lanes 0-15: K=k+0..7 / k+16..23,
    // lanes 16-31: K=k+8..15 / k+24..31 -> two contiguous 16B loads.
    const __bf16* Arow = A + (long)(m0 + l16) * K + half * 8;

    union Frag { v16bf v; uint4 u[2]; };

    for (int k0 = 0; k0 < K; k0 += GEMM_KC) {
        const int kc  = (K - k0 < GEMM_KC) ? (K - k0) : GEMM_KC;
        const int cu4 = kc >> 3;                        // uint4 per row of chunk

        __syncthreads();                                // previous chunk consumed
        // cooperative stage: rows n0..n0+63, K range [k0, k0+kc)
        for (int i = tid; i < 64 * cu4; i += 128) {
            int r = i / cu4, c = i - r * cu4;
            ((uint4*)sB)[r * (GEMM_LDSROW >> 3) + c] =
                *(const uint4*)(Bt + (long)(n0 + r) * K + k0 + c * 8);
        }
        __syncthreads();

        if (active) {
            for (int k = 0; k < kc; k += 32) {
                Frag af;
                af.u[0] = *(const uint4*)(Arow + k0 + k);
                af.u[1] = *(const uint4*)(Arow + k0 + k + 16);
                Frag bf[4];
                #pragma unroll
                for (int t = 0; t < 4; ++t) {
                    // B-frag (32x16): lane = col; lanes 0-15 K=k+0..7 / k+16..23,
                    // lanes 16-31 K=k+8..15 / k+24..31 -> two 16B LDS loads.
                    const __bf16* p = sB + (t * 16 + l16) * GEMM_LDSROW + k + half * 8;
                    bf[t].u[0] = *(const uint4*)p;
                    bf[t].u[1] = *(const uint4*)(p + 16);
                }
                #pragma unroll
                for (int t = 0; t < 4; ++t)
                    acc[t] = __builtin_amdgcn_wmma_f32_16x16x32_bf16(
                        false, af.v, false, bf[t].v, (short)0, acc[t], false, false);
            }
        }
    }

    if (active) {
        // C/D layout: VGPR r -> row m0 + half*8 + r, column n0 + t*16 + l16
        #pragma unroll
        for (int t = 0; t < 4; ++t) {
            const int col = n0 + t * 16 + l16;
            const float bval = doBias ? bias[col] : 0.0f;
            #pragma unroll
            for (int r = 0; r < 8; ++r) {
                float v = acc[t][r] + bval;
                if (doRelu) v = fmaxf(v, 0.0f);
                C[(long)(m0 + half * 8 + r) * N + col] = v;
            }
        }
    }
}

// ---------------------------------------------------------------------------
// GCN aggregation: vectorized self-loop init + wave-per-edge atomic scatter
// ---------------------------------------------------------------------------
__global__ void k_selfloop_init_v4(const float4* __restrict__ Hlin,
                                   const float* __restrict__ dis,
                                   float4* Agg, int N, int Dv) {
    int i = blockIdx.x * blockDim.x + threadIdx.x;
    if (i < N * Dv) {
        float d  = dis[i / Dv];
        float dd = d * d;
        float4 h = Hlin[i];
        Agg[i] = make_float4(h.x * dd, h.y * dd, h.z * dd, h.w * dd);
    }
}

template <int CH>   // floats per lane: 4 (D=128) or 2 (D=64)
__global__ void k_edge_scatter(const float* __restrict__ Hlin,
                               const long long* __restrict__ src,
                               const long long* __restrict__ tgt,
                               const float* __restrict__ dis,
                               float* Agg, int E) {
    int gid  = blockIdx.x * blockDim.x + threadIdx.x;
    int e    = gid >> 5;
    int lane = gid & 31;
    if (e >= E) return;
    int s = (int)src[e], t = (int)tgt[e];
    float w = dis[s] * dis[t];
    const int D = CH * 32;
    const float* hs = Hlin + (long)s * D + lane * CH;
    float*       at = Agg  + (long)t * D + lane * CH;
    if constexpr (CH == 4) {
        float4 h = *(const float4*)hs;
        atomicAdd(&at[0], h.x * w);
        atomicAdd(&at[1], h.y * w);
        atomicAdd(&at[2], h.z * w);
        atomicAdd(&at[3], h.w * w);
    } else {
        float2 h = *(const float2*)hs;
        atomicAdd(&at[0], h.x * w);
        atomicAdd(&at[1], h.y * w);
    }
}

__global__ void k_bias_relu_bf16(const float* __restrict__ Agg, const float* __restrict__ b,
                                 __bf16* out, int N, int D) {
    int i = blockIdx.x * blockDim.x + threadIdx.x;
    if (i < N * D) out[i] = (__bf16)fmaxf(Agg[i] + b[i % D], 0.0f);
}

// ---------------------------------------------------------------------------
// Mean pool (bias b2 folded in later), z assembly, fused BN + final dot
// ---------------------------------------------------------------------------
__global__ void k_pool(const float* __restrict__ Agg2, const long long* __restrict__ batch,
                       float* sums, float* cnt, int N, int D) {
    int gid  = blockIdx.x * blockDim.x + threadIdx.x;
    int i    = gid >> 5;
    int lane = gid & 31;
    if (i >= N) return;
    int g = (int)batch[i];
    for (int f = lane; f < D; f += 32)
        atomicAdd(&sums[g * D + f], Agg2[(long)i * D + f]);
    if (lane == 0) atomicAdd(&cnt[g], 1.0f);
}

__global__ void k_build_z(const float* __restrict__ sums, const float* __restrict__ cnt,
                          const float* __restrict__ b2, const float* __restrict__ img2,
                          __bf16* zbf, int G) {
    int gid = blockIdx.x * blockDim.x + threadIdx.x;
    if (gid >= G * 128) return;
    int g = gid >> 7, f = gid & 127;
    float v = (f < 64) ? (sums[g * 64 + f] / fmaxf(cnt[g], 1.0f) + b2[f])
                       : img2[g * 64 + (f - 64)];
    zbf[gid] = (__bf16)v;
}

__global__ void k_final(const float* __restrict__ z2,
                        const float* __restrict__ gamma, const float* __restrict__ beta,
                        const float* __restrict__ mean,  const float* __restrict__ var,
                        const float* __restrict__ Wc2,   const float* __restrict__ bc2,
                        float* out, int G) {
    int gid  = blockIdx.x * blockDim.x + threadIdx.x;
    int g    = gid >> 5;
    int lane = gid & 31;
    if (g >= G) return;
    float acc = 0.0f;
    for (int f = lane; f < 128; f += 32) {
        float z  = z2[g * 128 + f];
        float zn = (z - mean[f]) * rsqrtf(var[f] + 1e-5f) * gamma[f] + beta[f];
        acc += zn * Wc2[f];
    }
    #pragma unroll
    for (int off = 16; off > 0; off >>= 1) acc += __shfl_xor(acc, off, 32);
    if (lane == 0) out[g] = acc + bc2[0];
}

// ---------------------------------------------------------------------------
// Host launcher
// ---------------------------------------------------------------------------
static inline void launch_gemm(const __bf16* A, const __bf16* Bt, const float* bias,
                               float* C, int M, int K, int N, int doBias, int doRelu,
                               hipStream_t s) {
    int mtiles = (M + 15) / 16;
    dim3 grid((mtiles + 3) / 4, N / 64);
    k_gemm_bf16<<<grid, 128, 0, s>>>(A, Bt, bias, C, M, K, N, doBias, doRelu);
}

extern "C" void kernel_launch(void* const* d_in, const int* in_sizes, int n_in,
                              void* d_out, int out_size, void* d_ws, size_t ws_size,
                              hipStream_t stream) {
    (void)in_sizes; (void)n_in; (void)out_size; (void)ws_size;

    const float*     x     = (const float*)d_in[0];
    const long long* ei    = (const long long*)d_in[1];
    const long long* batch = (const long long*)d_in[2];
    const float*     imgf  = (const float*)d_in[3];
    const float* W1  = (const float*)d_in[4];  const float* b1  = (const float*)d_in[5];
    const float* W2  = (const float*)d_in[6];  const float* b2  = (const float*)d_in[7];
    const float* Wm1 = (const float*)d_in[8];  const float* bm1 = (const float*)d_in[9];
    const float* Wm2 = (const float*)d_in[10]; const float* bm2 = (const float*)d_in[11];
    const float* Wc1 = (const float*)d_in[12]; const float* bc1 = (const float*)d_in[13];
    const float* bng = (const float*)d_in[14]; const float* bnb = (const float*)d_in[15];
    const float* bnm = (const float*)d_in[16]; const float* bnv = (const float*)d_in[17];
    const float* Wc2 = (const float*)d_in[18]; const float* bc2 = (const float*)d_in[19];
    float* out = (float*)d_out;

    const int N = 100000, E = 1600000, F = 128, G = 256, IMG = 1280, GE = 64;
    const long long* src = ei;
    const long long* tgt = ei + E;

    char*  base = (char*)d_ws;
    size_t off  = 0;
    auto alloc = [&](size_t bytes) -> void* {
        void* p = base + off;
        off = (off + bytes + 255) & ~(size_t)255;
        return p;
    };
    float*  deg    = (float*)alloc((size_t)N * 4);
    float*  dis    = (float*)alloc((size_t)N * 4);
    __bf16* xbf    = (__bf16*)alloc((size_t)N * F * 2);
    __bf16* W1t    = (__bf16*)alloc((size_t)128 * 128 * 2);
    __bf16* W2t    = (__bf16*)alloc((size_t)64 * 128 * 2);
    __bf16* Wm1t   = (__bf16*)alloc((size_t)256 * 1280 * 2);
    __bf16* Wm2t   = (__bf16*)alloc((size_t)64 * 256 * 2);
    __bf16* Wc1t   = (__bf16*)alloc((size_t)128 * 128 * 2);
    float*  Hlin   = (float*)alloc((size_t)N * F * 4);   // reused: layer1 then layer2
    float*  Agg    = (float*)alloc((size_t)N * F * 4);   // reused: layer1 then layer2
    __bf16* H1bf   = (__bf16*)alloc((size_t)N * F * 2);
    __bf16* imgbf  = (__bf16*)alloc((size_t)G * IMG * 2);
    float*  img1   = (float*)alloc((size_t)G * 256 * 4);
    __bf16* img1bf = (__bf16*)alloc((size_t)G * 256 * 2);
    float*  img2   = (float*)alloc((size_t)G * GE * 4);
    float*  sums   = (float*)alloc((size_t)G * GE * 4);
    float*  cnt    = (float*)alloc((size_t)G * 4);
    __bf16* zbf    = (__bf16*)alloc((size_t)G * 128 * 2);
    float*  z2     = (float*)alloc((size_t)G * 128 * 4);

    auto cdiv = [](int a, int b) { return (a + b - 1) / b; };

    // --- degree / normalization ---
    k_fill_f32<<<cdiv(N, 256), 256, 0, stream>>>(deg, 1.0f, N);       // self-loop
    k_deg_atomic<<<cdiv(E, 256), 256, 0, stream>>>(tgt, deg, E);
    k_rsqrt<<<cdiv(N, 256), 256, 0, stream>>>(deg, dis, N);

    // --- precision conversion / weight transposes ---
    k_f32_to_bf16<<<cdiv(N * F, 256), 256, 0, stream>>>(x, xbf, N * F);
    k_f32_to_bf16<<<cdiv(G * IMG, 256), 256, 0, stream>>>(imgf, imgbf, G * IMG);
    k_transpose_bf16<<<cdiv(128 * 128, 256), 256, 0, stream>>>(W1, W1t, 128, 128);
    k_transpose_bf16<<<cdiv(128 * 64, 256), 256, 0, stream>>>(W2, W2t, 128, 64);
    k_transpose_bf16<<<cdiv(1280 * 256, 256), 256, 0, stream>>>(Wm1, Wm1t, 1280, 256);
    k_transpose_bf16<<<cdiv(256 * 64, 256), 256, 0, stream>>>(Wm2, Wm2t, 256, 64);
    k_transpose_bf16<<<cdiv(128 * 128, 256), 256, 0, stream>>>(Wc1, Wc1t, 128, 128);

    // --- GCN layer 1: x@W1 -> aggregate -> relu(+b1) ---
    launch_gemm(xbf, W1t, nullptr, Hlin, N, 128, 128, 0, 0, stream);
    k_selfloop_init_v4<<<cdiv(N * 32, 256), 256, 0, stream>>>(
        (const float4*)Hlin, dis, (float4*)Agg, N, 32);
    k_edge_scatter<4><<<cdiv(E * 32, 256), 256, 0, stream>>>(Hlin, src, tgt, dis, Agg, E);
    k_bias_relu_bf16<<<cdiv(N * 128, 256), 256, 0, stream>>>(Agg, b1, H1bf, N, 128);

    // --- GCN layer 2: h1@W2 -> aggregate (bias b2 folded into pooling) ---
    launch_gemm(H1bf, W2t, nullptr, Hlin, N, 128, 64, 0, 0, stream);
    k_selfloop_init_v4<<<cdiv(N * 16, 256), 256, 0, stream>>>(
        (const float4*)Hlin, dis, (float4*)Agg, N, 16);
    k_edge_scatter<2><<<cdiv(E * 32, 256), 256, 0, stream>>>(Hlin, src, tgt, dis, Agg, E);

    // --- global mean pool ---
    k_fill_f32<<<cdiv(G * GE, 256), 256, 0, stream>>>(sums, 0.0f, G * GE);
    k_fill_f32<<<1, 256, 0, stream>>>(cnt, 0.0f, G);
    k_pool<<<cdiv(N * 32, 256), 256, 0, stream>>>(Agg, batch, sums, cnt, N, 64);

    // --- image MLP ---
    launch_gemm(imgbf, Wm1t, bm1, img1, G, 1280, 256, 1, 1, stream);
    k_f32_to_bf16<<<cdiv(G * 256, 256), 256, 0, stream>>>(img1, img1bf, G * 256);
    launch_gemm(img1bf, Wm2t, bm2, img2, G, 256, 64, 1, 0, stream);

    // --- classifier: concat -> linear+relu -> BN -> final dot ---
    k_build_z<<<cdiv(G * 128, 256), 256, 0, stream>>>(sums, cnt, b2, img2, zbf, G);
    launch_gemm(zbf, Wc1t, bc1, z2, G, 128, 128, 1, 1, stream);
    k_final<<<cdiv(G * 32, 256), 256, 0, stream>>>(z2, bng, bnb, bnm, bnv, Wc2, bc2, out, G);
}
